// MinGRU_55362128445509
// MI455X (gfx1250) — compile-verified, matrix operand order checked by reading, not run on hardware
//
#include <hip/hip_runtime.h>

#define B_ 4
#define S_ 4096
#define D_ 1024
#define N2_ 2048            // 2*D
#define CHUNK 128
#define NCHUNK (S_ / CHUNK) // 32
#define NKSTEP (D_ / 32)    // 32 K-steps of 32

typedef __attribute__((ext_vector_type(16))) __bf16        v16bf;
typedef __attribute__((ext_vector_type(8)))  float         v8f;
typedef __attribute__((ext_vector_type(4)))  unsigned int  v4u;
typedef __attribute__((ext_vector_type(8)))  int           v8i;
typedef __attribute__((ext_vector_type(4)))  int           v4i;

#if __has_builtin(__builtin_amdgcn_tensor_load_to_lds)
#define HAVE_TDM 1
#else
#define HAVE_TDM 0
#endif

// ---------- helpers ----------

__device__ __forceinline__ unsigned short f2bf(float f) {
  // round-to-nearest-even fp32 -> bf16
  unsigned u = __builtin_bit_cast(unsigned, f);
  unsigned r = u + 0x7FFFu + ((u >> 16) & 1u);
  return (unsigned short)(r >> 16);
}

// 16x32 bf16 fragment from LDS [row][k] (row stride 32 bf16).
// elems 0..7 -> K = half*8+0..7 ; elems 8..15 -> K = 16+half*8+0..7
__device__ __forceinline__ v16bf frag_from_lds(const unsigned short* p) {
  union { uint4 q[2]; v16bf v; } u;
  u.q[0] = *(const uint4*)(p);
  u.q[1] = *(const uint4*)(p + 16);
  return u.v;
}

__device__ __forceinline__ float softplusf(float x) {
  return fmaxf(x, 0.f) + log1pf(expf(-fabsf(x)));
}
__device__ __forceinline__ float logaddexpf_(float a, float b) {
  float m = fmaxf(a, b);
  if (m == -__builtin_inff()) return -__builtin_inff();
  return m + log1pf(expf(fminf(a, b) - m));
}
__device__ __forceinline__ float log_gf(float x) {
  return (x >= 0.f) ? logf(x + 0.5f) : -softplusf(-x);
}
__device__ __forceinline__ void lc_lv(float hidden, float gate, float& lc, float& lv) {
  lc = -softplusf(gate);
  lv = -softplusf(-gate) + log_gf(hidden);
}

#if HAVE_TDM
// Issue a TDM 2-D tile load: global (bf16, row stride stride0 elements) -> LDS
// (rows stored contiguously). D# built per CDNA5 ISA 8.3-8.6. Wave-level op.
// This toolchain's builtin takes 6 args (g0,g1,g2,g3,extra,cpol).
__device__ __forceinline__ void tdm_load_2d(unsigned lds_off,
                                            const unsigned short* gsrc,
                                            unsigned tile_d0, unsigned tile_d1,
                                            unsigned tensor_d0, unsigned tensor_d1,
                                            unsigned stride0) {
  unsigned long long ga = (unsigned long long)(uintptr_t)gsrc;
  v4u g0;
  g0[0] = 1u;                                      // count=1, user mode
  g0[1] = lds_off;                                 // lds_addr
  g0[2] = (unsigned)(ga & 0xFFFFFFFFu);            // global_addr[31:0]
  g0[3] = (unsigned)((ga >> 32) & 0x01FFFFFFu)     // global_addr[56:32]
          | 0x80000000u;                           // type=2 ("image")
  v8i g1;
  g1[0] = 0x00010000;                              // wg_mask=0, data_size=1 (2B)
  g1[1] = (int)((tensor_d0 & 0xFFFFu) << 16);                      // dim0 lo16
  g1[2] = (int)((tensor_d0 >> 16) | ((tensor_d1 & 0xFFFFu) << 16));// dim0 hi/dim1 lo
  g1[3] = (int)((tensor_d1 >> 16) | (tile_d0 << 16));              // dim1 hi/tile0
  g1[4] = (int)tile_d1;                            // tile_dim1 (tile_dim2=0)
  g1[5] = (int)stride0;                            // dim0_stride[31:0]
  g1[6] = 0;                                       // dim0_stride hi / dim1_stride lo
  g1[7] = 0;                                       // dim1_stride hi
  v4i z4 = {0, 0, 0, 0};
  v8i z8 = {0, 0, 0, 0, 0, 0, 0, 0};
  __builtin_amdgcn_tensor_load_to_lds(g0, g1, z4, z4, z8, 0);
}
#endif

// ---------- pre-pass: fp32 -> bf16 ----------

__global__ __launch_bounds__(256) void convert_x_bf16(const float* __restrict__ x,
                                                      unsigned short* __restrict__ xbf) {
  const size_t i = ((size_t)blockIdx.x * 256 + threadIdx.x) * 8;
  float4 a = *(const float4*)(x + i);
  float4 b = *(const float4*)(x + i + 4);
  union { uint4 q; unsigned short u[8]; } o;
  o.u[0] = f2bf(a.x); o.u[1] = f2bf(a.y); o.u[2] = f2bf(a.z); o.u[3] = f2bf(a.w);
  o.u[4] = f2bf(b.x); o.u[5] = f2bf(b.y); o.u[6] = f2bf(b.z); o.u[7] = f2bf(b.w);
  *(uint4*)(xbf + i) = o.q;
}

// W (D x 2D, row major) -> Wt bf16 (2D x D): Wt[n][k] = bf16(W[k][n])
__global__ __launch_bounds__(256) void transpose_w_bf16(const float* __restrict__ W,
                                                        unsigned short* __restrict__ Wt) {
  __shared__ unsigned short tile[32][33];
  const int nt = blockIdx.x * 32;
  const int kt = blockIdx.y * 32;
  const int tx = threadIdx.x & 31;
  const int ty = threadIdx.x >> 5;  // 0..7
  for (int r = ty; r < 32; r += 8)
    tile[r][tx] = f2bf(W[(size_t)(kt + r) * N2_ + nt + tx]);  // coalesced over n
  __syncthreads();
  for (int r = ty; r < 32; r += 8)
    Wt[(size_t)(nt + r) * D_ + kt + tx] = tile[tx][r];        // coalesced over k
}

// ---------- GEMM: hg[m,n] = sum_k x[m,k]*W[k,n] ----------
// bf16 WMMA, f32 accum. Block tile 64(M) x 128(N), BK=32, 256 threads = 8 waves,
// each wave 32x32. LDS tiles double-buffered, filled by the Tensor Data Mover.

__global__ __launch_bounds__(256) void mingru_gemm_bf16(
    const unsigned short* __restrict__ xbf,  // [16384][1024] bf16 (m-major)
    const unsigned short* __restrict__ Wt,   // [2048][1024]  bf16 (n-major)
    float* __restrict__ hg) {
  __shared__ __align__(16) unsigned short sA[2][64 * 32];    // [m][k]
  __shared__ __align__(16) unsigned short sB[2][128 * 32];   // [n][k]

  const int tid  = threadIdx.x;
  const int lane = tid & 31;
  const int wave = tid >> 5;
  const int half = lane >> 4;
  const int l16  = lane & 15;

  const int bn = blockIdx.x * 128;
  const int bm = blockIdx.y * 64;
  const int waveM = (wave >> 2) * 32;  // 0 / 32
  const int waveN = (wave & 3) * 32;   // 0..96

  v8f acc[2][2];
#pragma unroll
  for (int i = 0; i < 2; ++i)
#pragma unroll
    for (int j = 0; j < 2; ++j)
#pragma unroll
      for (int v = 0; v < 8; ++v) acc[i][j][v] = 0.f;

#if HAVE_TDM
  // --- TDM double-buffered pipeline (wave 0 drives the DMA) ---
  if (wave == 0) {
    tdm_load_2d((unsigned)(uintptr_t)&sA[0][0], xbf + (size_t)bm * D_,
                32, 64, (unsigned)D_, (unsigned)(B_ * S_), (unsigned)D_);
    tdm_load_2d((unsigned)(uintptr_t)&sB[0][0], Wt + (size_t)bn * D_,
                32, 128, (unsigned)D_, (unsigned)N2_, (unsigned)D_);
  }
  for (int kt = 0; kt < NKSTEP; ++kt) {
    const int buf = kt & 1;
    if (wave == 0) {
      if (kt + 1 < NKSTEP) {
        const int k1 = (kt + 1) * 32;
        tdm_load_2d((unsigned)(uintptr_t)&sA[buf ^ 1][0],
                    xbf + (size_t)bm * D_ + k1,
                    32, 64, (unsigned)D_, (unsigned)(B_ * S_), (unsigned)D_);
        tdm_load_2d((unsigned)(uintptr_t)&sB[buf ^ 1][0],
                    Wt + (size_t)bn * D_ + k1,
                    32, 128, (unsigned)D_, (unsigned)N2_, (unsigned)D_);
        __builtin_amdgcn_s_wait_tensorcnt(2);  // stage kt fully landed in LDS
      } else {
        __builtin_amdgcn_s_wait_tensorcnt(0);
      }
    }
    __syncthreads();

    v16bf aF[2], bF[2];
#pragma unroll
    for (int i = 0; i < 2; ++i)
      aF[i] = frag_from_lds(&sA[buf][0] + (waveM + i * 16 + l16) * 32 + half * 8);
#pragma unroll
    for (int j = 0; j < 2; ++j)
      bF[j] = frag_from_lds(&sB[buf][0] + (waveN + j * 16 + l16) * 32 + half * 8);

#pragma unroll
    for (int i = 0; i < 2; ++i)
#pragma unroll
      for (int j = 0; j < 2; ++j)
        acc[i][j] = __builtin_amdgcn_wmma_f32_16x16x32_bf16(
            false, aF[i], false, bF[j], (short)0, acc[i][j], false, false);

    __syncthreads();  // everyone done reading buf before it is refilled
  }
#else
  // --- fallback: plain vectorized bf16 LDS copies, single buffer ---
  for (int kt = 0; kt < NKSTEP; ++kt) {
    const int k0 = kt * 32;
    __syncthreads();
    {
      const int row  = tid >> 2;
      const int col8 = (tid & 3) * 8;
      *(uint4*)&sA[0][row * 32 + col8] =
          *(const uint4*)(xbf + (size_t)(bm + row) * D_ + k0 + col8);
    }
    {
      const int n = tid >> 1;
      const int h = (tid & 1) * 16;
      uint4* dst = (uint4*)&sB[0][n * 32 + h];
      const uint4* src = (const uint4*)(Wt + (size_t)(bn + n) * D_ + k0 + h);
      dst[0] = src[0];
      dst[1] = src[1];
    }
    __syncthreads();

    v16bf aF[2], bF[2];
#pragma unroll
    for (int i = 0; i < 2; ++i)
      aF[i] = frag_from_lds(&sA[0][0] + (waveM + i * 16 + l16) * 32 + half * 8);
#pragma unroll
    for (int j = 0; j < 2; ++j)
      bF[j] = frag_from_lds(&sB[0][0] + (waveN + j * 16 + l16) * 32 + half * 8);

#pragma unroll
    for (int i = 0; i < 2; ++i)
#pragma unroll
      for (int j = 0; j < 2; ++j)
        acc[i][j] = __builtin_amdgcn_wmma_f32_16x16x32_bf16(
            false, aF[i], false, bF[j], (short)0, acc[i][j], false, false);
  }
#endif

  // C/D layout: VGPR v -> M = v + 8*half, N = lane&15
#pragma unroll
  for (int i = 0; i < 2; ++i) {
#pragma unroll
    for (int j = 0; j < 2; ++j) {
      const int gm0 = bm + waveM + i * 16 + half * 8;
      const int gn  = bn + waveN + j * 16 + l16;
#pragma unroll
      for (int v = 0; v < 8; ++v)
        hg[(size_t)(gm0 + v) * N2_ + gn] = acc[i][j][v];
    }
  }
}

// ---------- Pass 1: per-chunk segment summaries (LC, LV) ----------
__global__ __launch_bounds__(256) void mingru_chunk_summary(
    const float* __restrict__ hg, float* __restrict__ sumLC,
    float* __restrict__ sumLV) {
  const int tid = blockIdx.x * blockDim.x + threadIdx.x;  // 2^17 threads
  const int d = tid & (D_ - 1);
  const int c = (tid >> 10) & (NCHUNK - 1);
  const int b = tid >> 15;

  float LC = 0.f, LV = -__builtin_inff();
  const int s0 = c * CHUNK;
  for (int s = s0; s < s0 + CHUNK; ++s) {
    const size_t off = (size_t)(b * S_ + s) * N2_;
    float lc, lv;
    lc_lv(hg[off + d], hg[off + D_ + d], lc, lv);
    LC += lc;
    LV = logaddexpf_(lc + LV, lv);
  }
  const size_t idx = (size_t)(b * D_ + d) * NCHUNK + c;
  sumLC[idx] = LC;
  sumLV[idx] = LV;
}

// ---------- Pass 2: exclusive scan of chunk summaries per channel ----------
__global__ __launch_bounds__(256) void mingru_scan_chunks(
    const float* __restrict__ sumLC, const float* __restrict__ sumLV,
    float* __restrict__ chkIn) {
  const int tid = blockIdx.x * blockDim.x + threadIdx.x;  // B*D threads
  const size_t base = (size_t)tid * NCHUNK;
  float lh = -__builtin_inff();
  for (int c = 0; c < NCHUNK; ++c) {
    chkIn[base + c] = lh;
    lh = logaddexpf_(sumLC[base + c] + lh, sumLV[base + c]);
  }
}

// ---------- Pass 3: apply recurrence per chunk, write exp(log_h) ----------
__global__ __launch_bounds__(256) void mingru_apply(
    const float* __restrict__ hg, const float* __restrict__ chkIn,
    float* __restrict__ out, float* __restrict__ nph) {
  const int tid = blockIdx.x * blockDim.x + threadIdx.x;
  const int d = tid & (D_ - 1);
  const int c = (tid >> 10) & (NCHUNK - 1);
  const int b = tid >> 15;

  float lh = chkIn[(size_t)(b * D_ + d) * NCHUNK + c];
  const int s0 = c * CHUNK;
  for (int s = s0; s < s0 + CHUNK; ++s) {
    const size_t off = (size_t)(b * S_ + s) * N2_;
    float lc, lv;
    lc_lv(hg[off + d], hg[off + D_ + d], lc, lv);
    lh = logaddexpf_(lc + lh, lv);
    out[(size_t)(b * S_ + s) * D_ + d] = expf(lh);
  }
  if (c == NCHUNK - 1) nph[(size_t)b * D_ + d] = expf(lh);
}

// ---------- launch ----------
extern "C" void kernel_launch(void* const* d_in, const int* in_sizes, int n_in,
                              void* d_out, int out_size, void* d_ws, size_t ws_size,
                              hipStream_t stream) {
  (void)in_sizes; (void)n_in; (void)out_size; (void)ws_size;
  const float* x = (const float*)d_in[0];
  const float* W = (const float*)d_in[1];
  float* out = (float*)d_out;
  float* nph = out + (size_t)B_ * S_ * D_;

  char* ws = (char*)d_ws;
  float* hg = (float*)ws;                       ws += (size_t)B_ * S_ * N2_ * 4;   // 128 MB
  unsigned short* xbf = (unsigned short*)ws;    ws += (size_t)B_ * S_ * D_ * 2;    // 32 MB
  unsigned short* Wt  = (unsigned short*)ws;    ws += (size_t)N2_ * D_ * 2;        // 4 MB
  float* sumLC = (float*)ws;                    ws += (size_t)B_ * D_ * NCHUNK * 4;
  float* sumLV = (float*)ws;                    ws += (size_t)B_ * D_ * NCHUNK * 4;
  float* chkIn = (float*)ws;

  // pre-pass conversions
  convert_x_bf16<<<(B_ * S_ * D_) / (256 * 8), 256, 0, stream>>>(x, xbf);
  dim3 gT(N2_ / 32, D_ / 32);
  transpose_w_bf16<<<gT, 256, 0, stream>>>(W, Wt);

  // WMMA GEMM
  dim3 gGemm(N2_ / 128, (B_ * S_) / 64);  // (16, 256)
  mingru_gemm_bf16<<<gGemm, 256, 0, stream>>>(xbf, Wt, hg);

  // log-space chunked scan
  const int nScan = B_ * D_ * NCHUNK;  // 131072
  mingru_chunk_summary<<<nScan / 256, 256, 0, stream>>>(hg, sumLC, sumLV);
  mingru_scan_chunks<<<(B_ * D_) / 256, 256, 0, stream>>>(sumLC, sumLV, chkIn);
  mingru_apply<<<nScan / 256, 256, 0, stream>>>(hg, chkIn, out, nph);
}
